// GNNEncoder_1090921693880
// MI455X (gfx1250) — matrix operand order repaired and checked by reading, not compiled, not run on hardware
//
#include <hip/hip_runtime.h>
#include <hip/hip_bf16.h>
#include <stdint.h>

// GIN encoder for MI455X (gfx1250, wave32).
// Heavy GEMMs use v_wmma_f32_16x16x32_f16 (f16 A/B, f32 accumulate).
// Edge gather/scatter is the bandwidth bottleneck (~2 GB total @ 23.3 TB/s).

#define HD 128

typedef __attribute__((ext_vector_type(16))) _Float16 v16h;
typedef __attribute__((ext_vector_type(8)))  float    v8f;

union F16x2 { _Float16 h[2]; unsigned u; };
union Frag  { unsigned u[8]; uint4 q[2]; v16h v; };

// ---------------------------------------------------------------------------
// Pack an f32 weight matrix W[K][Nout] into WMMA B-fragment layout (f16 pairs).
// dst index = (((kb*nTiles + nt)*32 + lane)*8 + v); pair = (k, k+1) with
// k = kb*32 + 16*(v>>2) + 8*(lane>>4) + 2*(v&3), column n = nt*16 + (lane&15).
// ---------------------------------------------------------------------------
__global__ void pack_f16(const float* __restrict__ W, unsigned* __restrict__ dst,
                         int Nout, int nTiles, int total) {
    int idx = blockIdx.x * blockDim.x + threadIdx.x;
    if (idx >= total) return;
    int v    = idx & 7;
    int lane = (idx >> 3) & 31;
    int rest = idx >> 8;
    int nt   = rest % nTiles;
    int kb   = rest / nTiles;
    int k    = kb * 32 + 16 * (v >> 2) + 8 * (lane >> 4) + 2 * (v & 3);
    int n    = nt * 16 + (lane & 15);
    F16x2 p;
    p.h[0] = (_Float16)W[(size_t)k * Nout + n];
    p.h[1] = (_Float16)W[(size_t)(k + 1) * Nout + n];
    dst[idx] = p.u;
}

// ---------------------------------------------------------------------------
// Input projection: h = x @ in_W + in_b   (K=12 -> plain VALU, 32 lanes/node)
// ---------------------------------------------------------------------------
__global__ __launch_bounds__(256)
void in_proj(const float* __restrict__ x, const float* __restrict__ W,
             const float* __restrict__ b, float* __restrict__ h, int nNodes) {
    int idx  = blockIdx.x * 256 + threadIdx.x;
    int node = idx >> 5;
    if (node >= nNodes) return;
    int c0 = (idx & 31) * 4;
    float xi[12];
#pragma unroll
    for (int k = 0; k < 12; ++k) xi[k] = x[(size_t)node * 12 + k];
#pragma unroll
    for (int j = 0; j < 4; ++j) {
        int c = c0 + j;
        float acc = b[c];
#pragma unroll
        for (int k = 0; k < 12; ++k) acc += xi[k] * W[k * HD + c];
        h[(size_t)node * HD + c] = acc;
    }
}

// agg = (1 + eps_l) * h   (initialize scatter target)
__global__ __launch_bounds__(256)
void init_agg(const float* __restrict__ h, const float* __restrict__ eps_l,
              float* __restrict__ agg, int nNodes) {
    int idx = blockIdx.x * 256 + threadIdx.x;     // one float4 per thread
    if (idx >= nNodes * 32) return;
    float s = 1.0f + *eps_l;
    float4 v = *(const float4*)&h[(size_t)idx * 4];
    v.x *= s; v.y *= s; v.z *= s; v.w *= s;
    *(float4*)&agg[(size_t)idx * 4] = v;
}

// ---------------------------------------------------------------------------
// Edge pass: agg[dst] += h[src] + (edge_attr @ edge_W + edge_b)
// 32 lanes per edge, 4 channels per lane, float4 gather + f32 atomics.
// ---------------------------------------------------------------------------
__global__ __launch_bounds__(256)
void edge_kernel(const float* __restrict__ hin, const int* __restrict__ ei,
                 const float* __restrict__ ea, const float* __restrict__ eW,
                 const float* __restrict__ eb, float* __restrict__ agg, int nE) {
    int idx = blockIdx.x * 256 + threadIdx.x;
    int e = idx >> 5;
    if (e >= nE) return;
    int c0 = (idx & 31) * 4;
    int src = ei[e];
    int dst = ei[nE + e];
    float a0 = ea[(size_t)e * 3 + 0];
    float a1 = ea[(size_t)e * 3 + 1];
    float a2 = ea[(size_t)e * 3 + 2];
    float4 hv = *(const float4*)&hin[(size_t)src * HD + c0];
    const float* hp = (const float*)&hv;
#pragma unroll
    for (int j = 0; j < 4; ++j) {
        int c = c0 + j;
        float ev = eb[c] + a0 * eW[c] + a1 * eW[HD + c] + a2 * eW[2 * HD + c];
        atomicAdd(&agg[(size_t)dst * HD + c], hp[j] + ev);
    }
}

// ---------------------------------------------------------------------------
// GIN node MLP for one layer, 16-node tile per 128-thread block (4 waves):
//   z2 = relu(bn1(z @ W1 + b1));  y = relu(bn2(z2 @ W2 + b2));
//   h_out = LayerNorm(y + h_in)
// ---------------------------------------------------------------------------
__global__ __launch_bounds__(128)
void gin_mlp(const float* __restrict__ agg, const float* __restrict__ hin,
             float* __restrict__ hout,
             const unsigned* __restrict__ pW1,   // packed [4][16][32][8]
             const unsigned* __restrict__ pW2,   // packed [8][8][32][8]
             const float* __restrict__ b1, const float* __restrict__ g1,
             const float* __restrict__ bb1, const float* __restrict__ rm1,
             const float* __restrict__ rv1,
             const float* __restrict__ b2, const float* __restrict__ g2,
             const float* __restrict__ bb2, const float* __restrict__ rm2,
             const float* __restrict__ rv2,
             const float* __restrict__ lng, const float* __restrict__ lnb,
             int nNodes) {
    __shared__ float     zA[16][132];     // input tile (f32)
    __shared__ float     hres[16][132];   // residual tile
    __shared__ _Float16  z2[16][264];     // hidden (f16)
    __shared__ float     zO[16][132];     // pre-LN output
    __shared__ float     muS[16], rsS[16];

    const int tid   = threadIdx.x;
    const int lane  = tid & 31;
    const int wave  = tid >> 5;
    const int node0 = blockIdx.x * 16;
    const int half  = lane >> 4;
    const int mrow  = lane & 15;
    const int ncol  = lane & 15;

    // stage input + residual tiles (coalesced float4)
    for (int i = tid; i < 16 * 32; i += 128) {
        int r = i >> 5, c4 = (i & 31) * 4;
        int nr = node0 + r; if (nr >= nNodes) nr = nNodes - 1;
        *(float4*)&zA[r][c4]   = *(const float4*)&agg[(size_t)nr * HD + c4];
        *(float4*)&hres[r][c4] = *(const float4*)&hin[(size_t)nr * HD + c4];
    }
    __syncthreads();

    // ---- GEMM1: [16,128] @ [128,256], 4 n-tiles per wave ----
    v8f acc[4] = {};
    for (int kb = 0; kb < 4; ++kb) {
        Frag a;
#pragma unroll
        for (int v = 0; v < 8; ++v) {
            int k = kb * 32 + 16 * (v >> 2) + 8 * half + 2 * (v & 3);
            F16x2 p;
            p.h[0] = (_Float16)zA[mrow][k];
            p.h[1] = (_Float16)zA[mrow][k + 1];
            a.u[v] = p.u;
        }
#pragma unroll
        for (int t = 0; t < 4; ++t) {
            int nt = wave * 4 + t;
            Frag b;
            const uint4* src = (const uint4*)&pW1[(size_t)(((kb * 16 + nt) * 32) + lane) * 8];
            b.q[0] = src[0]; b.q[1] = src[1];
            acc[t] = __builtin_amdgcn_wmma_f32_16x16x32_f16(
                false, a.v, false, b.v, (short)0, acc[t], false, false);
        }
    }
#pragma unroll
    for (int t = 0; t < 4; ++t) {
        int n = (wave * 4 + t) * 16 + ncol;
        float sc  = g1[n] * rsqrtf(rv1[n] + 1e-5f);
        float off = (b1[n] - rm1[n]) * sc + bb1[n];
#pragma unroll
        for (int r = 0; r < 8; ++r) {
            float y = acc[t][r] * sc + off;
            y = y > 0.f ? y : 0.f;
            z2[r + 8 * half][n] = (_Float16)y;
        }
    }
    __syncthreads();

    // ---- GEMM2: [16,256] @ [256,128], 2 n-tiles per wave ----
    v8f acc2[2] = {};
    for (int kb = 0; kb < 8; ++kb) {
        Frag a;
#pragma unroll
        for (int v = 0; v < 8; ++v) {
            int k = kb * 32 + 16 * (v >> 2) + 8 * half + 2 * (v & 3);
            a.u[v] = *(const unsigned*)&z2[mrow][k];   // f16 pair, dword aligned
        }
#pragma unroll
        for (int t = 0; t < 2; ++t) {
            int nt = wave * 2 + t;
            Frag b;
            const uint4* src = (const uint4*)&pW2[(size_t)(((kb * 8 + nt) * 32) + lane) * 8];
            b.q[0] = src[0]; b.q[1] = src[1];
            acc2[t] = __builtin_amdgcn_wmma_f32_16x16x32_f16(
                false, a.v, false, b.v, (short)0, acc2[t], false, false);
        }
    }
#pragma unroll
    for (int t = 0; t < 2; ++t) {
        int n = (wave * 2 + t) * 16 + ncol;
        float sc  = g2[n] * rsqrtf(rv2[n] + 1e-5f);
        float off = (b2[n] - rm2[n]) * sc + bb2[n];
#pragma unroll
        for (int r = 0; r < 8; ++r) {
            int m = r + 8 * half;
            float y = acc2[t][r] * sc + off;
            y = y > 0.f ? y : 0.f;
            zO[m][n] = y + hres[m][n];
        }
    }
    __syncthreads();

    // ---- LayerNorm over 128 channels per row ----
    if (tid < 16) {
        float s = 0.f, s2 = 0.f;
        for (int c = 0; c < HD; ++c) { float v = zO[tid][c]; s += v; s2 += v * v; }
        float m = s * (1.0f / HD);
        float var = s2 * (1.0f / HD) - m * m;
        muS[tid] = m;
        rsS[tid] = rsqrtf(var + 1e-5f);
    }
    __syncthreads();
    for (int i = tid; i < 16 * HD; i += 128) {
        int r = i >> 7, c = i & 127;
        int nr = node0 + r;
        if (nr < nNodes)
            hout[(size_t)nr * HD + c] = (zO[r][c] - muS[r]) * rsS[r] * lng[c] + lnb[c];
    }
}

// ---------------------------------------------------------------------------
// Pool helpers
// ---------------------------------------------------------------------------
__global__ void zero_pool(float* gsum, float* gcnt) {
    int i = blockIdx.x * blockDim.x + threadIdx.x;
    if (i < 16 * HD) gsum[i] = 0.f;
    if (i < 16) gcnt[i] = 0.f;
}

__global__ __launch_bounds__(256)
void count_kernel(const int* __restrict__ batch, float* __restrict__ gcnt, int n) {
    __shared__ float hist[16];
    int t = threadIdx.x;
    if (t < 16) hist[t] = 0.f;
    __syncthreads();
    int i = blockIdx.x * 256 + t;
    if (i < n) atomicAdd(&hist[batch[i]], 1.0f);
    __syncthreads();
    if (t < 16 && hist[t] != 0.f) atomicAdd(&gcnt[t], hist[t]);
}

// ---------------------------------------------------------------------------
// Readout: node_emb = relu(cat @ out_W1 + b1) @ out_W2 + b2, plus segmented
// pooling (batch is sorted -> few atomic flushes per column).
// ---------------------------------------------------------------------------
__global__ __launch_bounds__(128)
void out_kernel(const float* __restrict__ h0, const float* __restrict__ h1,
                const float* __restrict__ h2, const float* __restrict__ h3,
                const unsigned* __restrict__ pO1,   // packed [16][8][32][8]
                const unsigned* __restrict__ pO2,   // packed [4][8][32][8]
                const float* __restrict__ ob1, const float* __restrict__ ob2,
                const int* __restrict__ batch, float* __restrict__ node_out,
                float* __restrict__ gsum, int nNodes) {
    __shared__ float    catA[16][516];   // [16,512] concat tile
    __shared__ _Float16 z2h[16][136];
    __shared__ float    zO[16][132];
    __shared__ int      bb[16];

    const int tid   = threadIdx.x;
    const int lane  = tid & 31;
    const int wave  = tid >> 5;
    const int node0 = blockIdx.x * 16;
    const int half  = lane >> 4;
    const int mrow  = lane & 15;
    const int ncol  = lane & 15;

    const float* hsrc[4] = { h0, h1, h2, h3 };
#pragma unroll
    for (int bsel = 0; bsel < 4; ++bsel) {
        const float* hb = hsrc[bsel];
        for (int i = tid; i < 16 * 32; i += 128) {
            int r = i >> 5, c4 = (i & 31) * 4;
            int nr = node0 + r; if (nr >= nNodes) nr = nNodes - 1;
            *(float4*)&catA[r][bsel * HD + c4] = *(const float4*)&hb[(size_t)nr * HD + c4];
        }
    }
    if (tid < 16) {
        int nr = node0 + tid; if (nr >= nNodes) nr = nNodes - 1;
        bb[tid] = batch[nr];
    }
    __syncthreads();

    // ---- GEMM: [16,512] @ [512,128], 2 n-tiles per wave ----
    v8f acc[2] = {};
    for (int kb = 0; kb < 16; ++kb) {
        Frag a;
#pragma unroll
        for (int v = 0; v < 8; ++v) {
            int k = kb * 32 + 16 * (v >> 2) + 8 * half + 2 * (v & 3);
            F16x2 p;
            p.h[0] = (_Float16)catA[mrow][k];
            p.h[1] = (_Float16)catA[mrow][k + 1];
            a.u[v] = p.u;
        }
#pragma unroll
        for (int t = 0; t < 2; ++t) {
            int nt = wave * 2 + t;
            Frag b;
            const uint4* src = (const uint4*)&pO1[(size_t)(((kb * 8 + nt) * 32) + lane) * 8];
            b.q[0] = src[0]; b.q[1] = src[1];
            acc[t] = __builtin_amdgcn_wmma_f32_16x16x32_f16(
                false, a.v, false, b.v, (short)0, acc[t], false, false);
        }
    }
#pragma unroll
    for (int t = 0; t < 2; ++t) {
        int n = (wave * 2 + t) * 16 + ncol;
        float bias = ob1[n];
#pragma unroll
        for (int r = 0; r < 8; ++r) {
            float y = acc[t][r] + bias;
            y = y > 0.f ? y : 0.f;
            z2h[r + 8 * half][n] = (_Float16)y;
        }
    }
    __syncthreads();

    // ---- GEMM: [16,128] @ [128,128] ----
    v8f acc2[2] = {};
    for (int kb = 0; kb < 4; ++kb) {
        Frag a;
#pragma unroll
        for (int v = 0; v < 8; ++v) {
            int k = kb * 32 + 16 * (v >> 2) + 8 * half + 2 * (v & 3);
            a.u[v] = *(const unsigned*)&z2h[mrow][k];
        }
#pragma unroll
        for (int t = 0; t < 2; ++t) {
            int nt = wave * 2 + t;
            Frag b;
            const uint4* src = (const uint4*)&pO2[(size_t)(((kb * 8 + nt) * 32) + lane) * 8];
            b.q[0] = src[0]; b.q[1] = src[1];
            acc2[t] = __builtin_amdgcn_wmma_f32_16x16x32_f16(
                false, a.v, false, b.v, (short)0, acc2[t], false, false);
        }
    }
#pragma unroll
    for (int t = 0; t < 2; ++t) {
        int n = (wave * 2 + t) * 16 + ncol;
        float bias = ob2[n];
#pragma unroll
        for (int r = 0; r < 8; ++r)
            zO[r + 8 * half][n] = acc2[t][r] + bias;
    }
    __syncthreads();

    // write node embeddings (coalesced) + segmented pooling per column
    {
        int c = tid;                 // 128 columns, blockDim == 128
        float s = 0.f;
        int cur = bb[0];
        for (int r = 0; r < 16; ++r) {
            int nr = node0 + r;
            if (nr >= nNodes) break;
            float v = zO[r][c];
            node_out[(size_t)nr * HD + c] = v;
            if (bb[r] != cur) { atomicAdd(&gsum[cur * HD + c], s); s = 0.f; cur = bb[r]; }
            s += v;
        }
        atomicAdd(&gsum[cur * HD + c], s);
    }
}

__global__ void finalize(const float* __restrict__ gsum, const float* __restrict__ gcnt,
                         float* __restrict__ gout) {
    int i = blockIdx.x * blockDim.x + threadIdx.x;
    if (i >= 16 * HD) return;
    float c = gcnt[i >> 7];
    if (c < 1.f) c = 1.f;
    gout[i] = gsum[i] / c;
}

// ---------------------------------------------------------------------------
extern "C" void kernel_launch(void* const* d_in, const int* in_sizes, int n_in,
                              void* d_out, int out_size, void* d_ws, size_t ws_size,
                              hipStream_t stream) {
    const float* x      = (const float*)d_in[0];
    const int*   ei     = (const int*)  d_in[1];
    const float* ea     = (const float*)d_in[2];
    const int*   batch  = (const int*)  d_in[3];
    const float* in_W   = (const float*)d_in[4];
    const float* in_b   = (const float*)d_in[5];
    const float* edge_W = (const float*)d_in[6];
    const float* edge_b = (const float*)d_in[7];
    const float* mlp_W1 = (const float*)d_in[8];
    const float* mlp_b1 = (const float*)d_in[9];
    const float* bn1_g  = (const float*)d_in[10];
    const float* bn1_b  = (const float*)d_in[11];
    const float* bn1_rm = (const float*)d_in[12];
    const float* bn1_rv = (const float*)d_in[13];
    const float* mlp_W2 = (const float*)d_in[14];
    const float* mlp_b2 = (const float*)d_in[15];
    const float* bn2_g  = (const float*)d_in[16];
    const float* bn2_b  = (const float*)d_in[17];
    const float* bn2_rm = (const float*)d_in[18];
    const float* bn2_rv = (const float*)d_in[19];
    const float* eps    = (const float*)d_in[20];
    const float* ln_g   = (const float*)d_in[21];
    const float* ln_b   = (const float*)d_in[22];
    const float* out_W1 = (const float*)d_in[23];
    const float* out_b1 = (const float*)d_in[24];
    const float* out_W2 = (const float*)d_in[25];
    const float* out_b2 = (const float*)d_in[26];

    const int N = in_sizes[0] / 12;
    const int E = in_sizes[1] / 2;
    const size_t NH = (size_t)N * HD;

    // workspace layout (~257 MB)
    float* ws = (float*)d_ws;
    float* hb0  = ws;
    float* hb1  = ws + NH;
    float* hb2  = ws + 2 * NH;
    float* hb3  = ws + 3 * NH;
    float* agg  = ws + 4 * NH;
    float* gsum = ws + 5 * NH;
    float* gcnt = gsum + 16 * HD;
    unsigned* pW1 = (unsigned*)(gcnt + 16);      // 3 * 16384
    unsigned* pW2 = pW1 + 3 * 16384;             // 3 * 16384
    unsigned* pO1 = pW2 + 3 * 16384;             // 32768
    unsigned* pO2 = pO1 + 32768;                 // 8192
    float* hbuf[4] = { hb0, hb1, hb2, hb3 };

    // pack weights into WMMA fragment layout (f16)
    for (int l = 0; l < 3; ++l) {
        pack_f16<<<16384 / 256, 256, 0, stream>>>(mlp_W1 + (size_t)l * 128 * 256,
                                                  pW1 + l * 16384, 256, 16, 16384);
        pack_f16<<<16384 / 256, 256, 0, stream>>>(mlp_W2 + (size_t)l * 256 * 128,
                                                  pW2 + l * 16384, 128, 8, 16384);
    }
    pack_f16<<<32768 / 256, 256, 0, stream>>>(out_W1, pO1, 128, 8, 32768);
    pack_f16<<<8192 / 256, 256, 0, stream>>>(out_W2, pO2, 128, 8, 8192);

    // input projection
    in_proj<<<(N * 32 + 255) / 256, 256, 0, stream>>>(x, in_W, in_b, hbuf[0], N);

    const int nTiles = (N + 15) / 16;
    for (int l = 0; l < 3; ++l) {
        init_agg<<<(N * 32 + 255) / 256, 256, 0, stream>>>(hbuf[l], eps + l, agg, N);
        edge_kernel<<<(E * 32 + 255) / 256, 256, 0, stream>>>(
            hbuf[l], ei, ea, edge_W + (size_t)l * 3 * HD, edge_b + l * HD, agg, E);
        gin_mlp<<<nTiles, 128, 0, stream>>>(
            agg, hbuf[l], hbuf[l + 1],
            pW1 + l * 16384, pW2 + l * 16384,
            mlp_b1 + l * 256, bn1_g + l * 256, bn1_b + l * 256,
            bn1_rm + l * 256, bn1_rv + l * 256,
            mlp_b2 + l * 128, bn2_g + l * 128, bn2_b + l * 128,
            bn2_rm + l * 128, bn2_rv + l * 128,
            ln_g + l * 128, ln_b + l * 128, N);
    }

    zero_pool<<<(16 * HD + 255) / 256, 256, 0, stream>>>(gsum, gcnt);
    count_kernel<<<(N + 255) / 256, 256, 0, stream>>>(batch, gcnt, N);
    out_kernel<<<nTiles, 128, 0, stream>>>(hbuf[0], hbuf[1], hbuf[2], hbuf[3],
                                           pO1, pO2, out_b1, out_b2, batch,
                                           (float*)d_out, gsum, N);
    finalize<<<(16 * HD + 255) / 256, 256, 0, stream>>>(gsum, gcnt, (float*)d_out + NH);
}
